// MHPECCA_51754355917176
// MI455X (gfx1250) — matrix operand order repaired and checked by reading, not compiled, not run on hardware
//
#include <hip/hip_runtime.h>
#include <hip/hip_bf16.h>
#include <math.h>

// ---------------- problem constants ----------------
#define B_   8
#define C_   512
#define NH_  4
#define H_   64
#define W_   64
#define CH_  64      // qkv channels
#define HC_  16      // head channels
#define HW_  4096    // H_*W_

typedef __bf16 bf16_t;
typedef __attribute__((ext_vector_type(16))) bf16_t v16bf;
typedef __attribute__((ext_vector_type(8)))  float  v8f;

union pack8 { uint4 u; bf16_t h[8]; };
union packf4 { float4 f; float s[4]; };

// ---------------- WMMA helpers (CDNA5 wave32) ----------------
static __device__ __forceinline__ v8f zero8() {
  v8f z;
#pragma unroll
  for (int i = 0; i < 8; ++i) z[i] = 0.0f;
  return z;
}

static __device__ __forceinline__ v8f wmma_bf16(v16bf a, v16bf b, v8f c) {
  // D = A(16x32) * B(32x16) + C, f32 accumulate
  return __builtin_amdgcn_wmma_f32_16x16x32_bf16(false, a, false, b, (short)0, c,
                                                 false, false);
}

// A fragment: 16x32 bf16 from row-major (M-major) source, leading dim `ld`.
// One contiguous 32B run per lane (k = 16*(lane/16) + s) -> 2x ds_load_b128.
static __device__ __forceinline__ v16bf load_frag_a(const bf16_t* __restrict__ src, int ld) {
  const int lane = threadIdx.x & 31;
  const int m  = lane & 15;
  const int kb = (lane >> 4) << 4;
  v16bf a;
#pragma unroll
  for (int j = 0; j < 8; ++j) {
    int k = kb + 2 * j;
    a[2 * j]     = src[m * ld + k];
    a[2 * j + 1] = src[m * ld + k + 1];
  }
  return a;
}

// B fragment: 32x16 bf16 from an N-MAJOR (transposed) source: src[n*ld + k].
// One contiguous 32B run per lane -> 2x ds_load_b128.
static __device__ __forceinline__ v16bf load_frag_bt(const bf16_t* __restrict__ src, int ld) {
  const int lane = threadIdx.x & 31;
  const int n  = lane & 15;
  const int kb = (lane >> 4) << 4;
  v16bf b;
#pragma unroll
  for (int j = 0; j < 16; ++j) b[j] = src[n * ld + kb + j];
  return b;
}

// ---------------- kernel 1: positional encoding (torch raw-view faithful) ----
__global__ void cca_pos_kernel(float* __restrict__ pos) {
  int idx = blockIdx.x * blockDim.x + threadIdx.x;
  if (idx >= C_ * HW_) return;
  int c   = idx >> 12;
  int rem = idx & 4095;
  int hh  = rem >> 6;
  int ww  = rem & 63;
  const float kfac = 9.210340371976184f / (float)C_;   // ln(10000)/C
  // pos_h: (h, c) raw-viewed as (c, h)
  int f1  = c * H_ + hh;
  int row = f1 >> 9;            // / C_
  int col = f1 & (C_ - 1);
  int j1  = (col < C_ / 2) ? col : (col - C_ / 2);
  float d1 = __expf(-(float)(2 * j1) * kfac);
  float vh = (col < C_ / 2) ? __sinf((float)row * d1) : __cosf((float)row * d1);
  // pos_w: (c/2, 2w) raw-viewed as (c, w)
  int f2   = c * W_ + ww;
  int row2 = f2 >> 7;           // / (2W)
  int col2 = f2 & (2 * W_ - 1);
  float d2 = __expf(-(float)(2 * row2) * kfac);
  float vw = (col2 < W_) ? __sinf((float)col2 * d2) : __cosf((float)(col2 - W_) * d2);
  pos[idx] = vh + vw;
}

// ---------------- kernel 2: weights -> bf16 ---------------------------------
__global__ void cca_wcvt_kernel(const float* __restrict__ wq, const float* __restrict__ wk,
                                const float* __restrict__ wv, const float* __restrict__ wo,
                                bf16_t* __restrict__ wqkvb, bf16_t* __restrict__ wob) {
  int idx = blockIdx.x * blockDim.x + threadIdx.x;
  const int NQKV = 3 * CH_ * C_;   // 98304
  const int NWO  = C_ * CH_;       // 32768
  if (idx < NQKV) {
    int which = idx / (CH_ * C_);
    int r     = idx % (CH_ * C_);
    const float* src = (which == 0) ? wq : ((which == 1) ? wk : wv);
    wqkvb[idx] = (bf16_t)src[r];
  } else if (idx < NQKV + NWO) {
    wob[idx - NQKV] = (bf16_t)wo[idx - NQKV];
  }
}

// ---------------- kernel 3: QKV GEMM (M=192, K=512, N=4096/batch) -----------
// block: 64 output columns of one batch; 8 waves x 6 WMMA tiles.
// nt = (wave+8s)&3 is wave-constant -> one B fragment reused by 6 WMMAs.
// The (x + pos) -> bf16 conversion is fused into B staging: x is consumed by
// exactly one block, so no xp_bf16 intermediate is needed (saves ~64MB HBM).
#define LDA_ 40   // padded leading dim for 32-wide K tiles (16B aligned, bank-safe)
__global__ void __launch_bounds__(256) cca_qkv_kernel(
    const float* __restrict__ x, const float* __restrict__ pos,
    const bf16_t* __restrict__ wqkvb,
    const float* __restrict__ bq, const float* __restrict__ bk,
    const float* __restrict__ bv, bf16_t* __restrict__ qkv) {
  __shared__ alignas(16) bf16_t As[192 * LDA_];   // M-major: As[m][k]
  __shared__ alignas(16) bf16_t BsT[64 * LDA_];   // N-major: BsT[n][k]
  const int tid  = threadIdx.x;
  const int wave = tid >> 5;
  const int lane = tid & 31;
  const int b    = blockIdx.x >> 6;
  const int nblk = blockIdx.x & 63;
  const float* xb = x + (size_t)b * C_ * HW_;

  v8f acc[6];
#pragma unroll
  for (int s = 0; s < 6; ++s) acc[s] = zero8();

  for (int k0 = 0; k0 < C_; k0 += 32) {
    __syncthreads();
    // A: 192x32 = 768 8-elem chunks; vectorized b128 global->LDS
    for (int ch = tid; ch < 768; ch += 256) {
      int m = ch >> 2, ko = (ch & 3) * 8;
      uint4 d = *(const uint4*)(wqkvb + m * C_ + k0 + ko);
      *(uint4*)(As + m * LDA_ + ko) = d;
    }
    // B: 32(k) x 64(n); each thread: 8 n-values. Coalesced float4 reads of
    // x and pos, fused add + bf16 convert, transposed scatter into LDS.
    {
      int k = tid >> 3, ng = (tid & 7) * 8;
      size_t goff = (size_t)(k0 + k) * HW_ + nblk * 64 + ng;
      packf4 x0, x1, p0, p1;
      x0.f = *(const float4*)(xb + goff);
      x1.f = *(const float4*)(xb + goff + 4);
      p0.f = *(const float4*)(pos + goff);
      p1.f = *(const float4*)(pos + goff + 4);
#pragma unroll
      for (int i = 0; i < 4; ++i) {
        BsT[(ng + i) * LDA_ + k]     = (bf16_t)(x0.s[i] + p0.s[i]);
        BsT[(ng + 4 + i) * LDA_ + k] = (bf16_t)(x1.s[i] + p1.s[i]);
      }
    }
    if (k0 + 32 < C_) {  // prefetch next x tile (global_prefetch_b8)
      __builtin_prefetch(xb + (size_t)(k0 + 32 + (tid >> 3)) * HW_ + nblk * 64 + (tid & 7) * 8, 0, 1);
    }
    __syncthreads();
#pragma unroll
    for (int s = 0; s < 6; ++s) {
      int t  = wave + 8 * s;
      int mt = t >> 2, nt = t & 3;           // nt == wave&3: constant per wave
      v16bf a  = load_frag_a(As + mt * 16 * LDA_, LDA_);
      v16bf bb = load_frag_bt(BsT + nt * 16 * LDA_, LDA_);
      acc[s] = wmma_bf16(a, bb, acc[s]);
    }
  }

  const int n_  = lane & 15;
  const int mb8 = (lane >> 4) << 3;
#pragma unroll
  for (int s = 0; s < 6; ++s) {
    int t  = wave + 8 * s;
    int mt = t >> 2, nt = t & 3;
#pragma unroll
    for (int r = 0; r < 8; ++r) {
      int m    = mt * 16 + mb8 + r;
      int col  = nblk * 64 + nt * 16 + n_;
      float bias = (m < 64) ? bq[m] : ((m < 128) ? bk[m - 64] : bv[m - 128]);
      qkv[((size_t)b * 192 + m) * HW_ + col] = (bf16_t)(acc[s][r] + bias);
    }
  }
}

// ---------------- kernel 4: fused criss-cross attention ---------------------
// one block per (b, head, line u): energy (64x128) -> softmax -> out_H/out_W.
#define LDV_  72    // padded leading dim for 64-wide K tiles
#define LDE_  132   // padded float stride for 128-wide logit rows
#define LDT_  136   // padded bf16 stride for 128-wide att rows (16B aligned)
__global__ void __launch_bounds__(256) cca_attn_kernel(
    const bf16_t* __restrict__ qkv, bf16_t* __restrict__ outH, bf16_t* __restrict__ outW) {
  __shared__ alignas(16) bf16_t qcT[64 * LDA_];   // A: q column view, qcT[i][c] (K padded)
  __shared__ alignas(16) bf16_t qrT[64 * LDA_];   // A: q row view
  __shared__ alignas(16) bf16_t kcT[64 * LDA_];   // B (N-major): kcT[j][c]
  __shared__ alignas(16) bf16_t krT[64 * LDA_];
  __shared__ alignas(16) bf16_t vcs[16 * LDV_];   // A: vcs[c][j]
  __shared__ alignas(16) bf16_t vrs[16 * LDV_];
  __shared__ alignas(16) float  Ebuf[64 * LDE_];
  __shared__ alignas(16) bf16_t attr[64 * LDT_];  // att row-major: attr[i][j], j=0..127
  __shared__ float  pmax[4 * 64];
  __shared__ float  psum[4 * 64];

  const int tid  = threadIdx.x;
  const int wave = tid >> 5;
  const int lane = tid & 31;
  const int b    = blockIdx.x >> 8;
  const int n    = (blockIdx.x >> 6) & 3;
  const int u    = blockIdx.x & 63;

  const size_t qb = ((size_t)b * 192 + n * HC_) * HW_;
  const size_t kb = ((size_t)b * 192 + 64 + n * HC_) * HW_;
  const size_t vb = ((size_t)b * 192 + 128 + n * HC_) * HW_;

  // ---- row views: contiguous in i -> coalesced b128 loads, LDS scatter ----
  if (tid < 128) {
    int c = tid >> 3, ig = (tid & 7) * 8;
    pack8 dq, dk, dv;
    dq.u = *(const uint4*)(qkv + qb + (size_t)c * HW_ + u * 64 + ig);
    dk.u = *(const uint4*)(qkv + kb + (size_t)c * HW_ + u * 64 + ig);
    dv.u = *(const uint4*)(qkv + vb + (size_t)c * HW_ + u * 64 + ig);
#pragma unroll
    for (int i = 0; i < 8; ++i) {
      qrT[(ig + i) * LDA_ + c] = dq.h[i];
      krT[(ig + i) * LDA_ + c] = dk.h[i];
    }
    *(uint4*)(vrs + c * LDV_ + ig) = dv.u;   // vrs row is contiguous in j
  }
  // ---- column views: strided gathers, i fastest (one line per lane) ----
  for (int e = tid; e < 16 * 64; e += 256) {
    int c = e >> 6, i = e & 63;
    size_t off = (size_t)c * HW_ + i * 64 + u;
    qcT[i * LDA_ + c] = qkv[qb + off];
    kcT[i * LDA_ + c] = qkv[kb + off];
    vcs[c * LDV_ + i] = qkv[vb + off];
  }
  // ---- zero the K padding (channels 16..31) ----
  for (int e = tid; e < 64 * 16; e += 256) {
    int i = e >> 4, c = 16 + (e & 15);
    bf16_t z = (bf16_t)0.0f;
    qcT[i * LDA_ + c] = z;
    qrT[i * LDA_ + c] = z;
    kcT[i * LDA_ + c] = z;
    krT[i * LDA_ + c] = z;
  }
  __syncthreads();

  // ---- energy: 4 i-tiles x 8 j-tiles = 32 WMMA tiles, 4 per wave ----
  {
    const int n_  = lane & 15;
    const int mb8 = (lane >> 4) << 3;
#pragma unroll
    for (int s = 0; s < 4; ++s) {
      int t  = wave + 8 * s;
      int it = t & 3, jt = t >> 2;
      const bf16_t* asrc = ((jt < 4) ? qcT : qrT) + it * 16 * LDA_;
      const bf16_t* bsrc = ((jt < 4) ? kcT : krT) + (jt & 3) * 16 * LDA_;
      v16bf a  = load_frag_a(asrc, LDA_);
      v16bf bb = load_frag_bt(bsrc, LDA_);
      v8f d = wmma_bf16(a, bb, zero8());
#pragma unroll
      for (int r = 0; r < 8; ++r) {
        int i = it * 16 + mb8 + r;
        int j = jt * 16 + n_;            // 0..127 (H part then W part)
        float val = d[r];
        if (jt < 4 && i == j) val = -3.0e38f;   // -inf diagonal mask on H part
        Ebuf[i * LDE_ + j] = val;
      }
    }
  }
  __syncthreads();

  // ---- softmax over 128 logits per row (4 threads per row) ----
  {
    int r = tid & 63, p = tid >> 6;
    float m = -3.4e38f;
    for (int jj = 0; jj < 32; ++jj) m = fmaxf(m, Ebuf[r * LDE_ + p * 32 + jj]);
    pmax[p * 64 + r] = m;
    __syncthreads();
    float rm = fmaxf(fmaxf(pmax[r], pmax[64 + r]), fmaxf(pmax[128 + r], pmax[192 + r]));
    float s = 0.0f;
    for (int jj = 0; jj < 32; ++jj) {
      float e = __expf(Ebuf[r * LDE_ + p * 32 + jj] - rm);
      Ebuf[r * LDE_ + p * 32 + jj] = e;
      s += e;
    }
    psum[p * 64 + r] = s;
    __syncthreads();
    float inv = 1.0f / (psum[r] + psum[64 + r] + psum[128 + r] + psum[192 + r]);
    for (int jj = 0; jj < 32; ++jj) {
      int j = p * 32 + jj;
      attr[r * LDT_ + j] = (bf16_t)(Ebuf[r * LDE_ + j] * inv);  // row-major store
    }
  }
  __syncthreads();

  // ---- aggregation: out = V(16x64) @ att^T(64x64); 8 tiles, 1 per wave ----
  // out[c][i] = sum_j V[c][j] * att[i][j]; att row-major == B in N-major form.
  {
    int half = wave >> 2, nt = wave & 3;
    const bf16_t* vsrc = half ? vrs : vcs;
    v8f acc = zero8();
#pragma unroll
    for (int ks = 0; ks < 2; ++ks) {
      v16bf a  = load_frag_a(vsrc + ks * 32, LDV_);
      v16bf bb = load_frag_bt(attr + nt * 16 * LDT_ + half * 64 + ks * 32, LDT_);
      acc = wmma_bf16(a, bb, acc);
    }
    const int n_  = lane & 15;
    const int mb8 = (lane >> 4) << 3;
#pragma unroll
    for (int r = 0; r < 8; ++r) {
      int cch = mb8 + r;             // head channel 0..15
      int i   = nt * 16 + n_;        // position along the attended line
      if (half == 0)
        outH[(((size_t)b * 64 + n * 16 + cch) * 64 + i) * 64 + u] = (bf16_t)acc[r];
      else
        outW[(((size_t)b * 64 + n * 16 + cch) * 64 + u) * 64 + i] = (bf16_t)acc[r];
    }
  }
}

// ---------------- kernel 5: output conv + residual ---------------------------
// out = gamma * (Wo(512x64) @ (outH+outW) + bo) + x + pos
__global__ void __launch_bounds__(256) cca_out_kernel(
    const bf16_t* __restrict__ outH, const bf16_t* __restrict__ outW,
    const bf16_t* __restrict__ wob, const float* __restrict__ bo,
    const float* __restrict__ x, const float* __restrict__ pos,
    const float* __restrict__ gamma, float* __restrict__ out) {
  __shared__ alignas(16) bf16_t Atile[64 * LDV_];   // Wo tile, M-major (K=64)
  __shared__ alignas(16) bf16_t BtileT[64 * LDV_];  // (outH+outW) tile, N-major
  const int tid  = threadIdx.x;
  const int wave = tid >> 5;
  const int lane = tid & 31;
  const int b    = blockIdx.x >> 9;
  const int mtb  = (blockIdx.x >> 6) & 7;
  const int nb   = blockIdx.x & 63;
  const float g  = gamma[0];

  // A: vectorized b128 copy (rows contiguous in K)
  for (int ch = tid; ch < 512; ch += 256) {
    int r = ch >> 3, ko = (ch & 7) * 8;
    uint4 d = *(const uint4*)(wob + (mtb * 64 + r) * 64 + ko);
    *(uint4*)(Atile + r * LDV_ + ko) = d;
  }
  // B: b128 coalesced reads of bf16 outH/outW, add, transpose into N-major LDS
  for (int ch = tid; ch < 512; ch += 256) {
    int r = ch >> 3, cg = (ch & 7) * 8;         // r = K channel, cg = spatial col
    size_t idx = ((size_t)b * 64 + r) * HW_ + nb * 64 + cg;
    pack8 hv, wv;
    hv.u = *(const uint4*)(outH + idx);
    wv.u = *(const uint4*)(outW + idx);
#pragma unroll
    for (int i = 0; i < 8; ++i)
      BtileT[(cg + i) * LDV_ + r] = (bf16_t)((float)hv.h[i] + (float)wv.h[i]);
  }
  __syncthreads();

  v8f acc[2];
  acc[0] = zero8(); acc[1] = zero8();
#pragma unroll
  for (int s = 0; s < 2; ++s) {
    int t  = wave + 8 * s;
    int mt = t >> 2, nt = t & 3;
#pragma unroll
    for (int ks = 0; ks < 2; ++ks) {
      v16bf a  = load_frag_a(Atile + mt * 16 * LDV_ + ks * 32, LDV_);
      v16bf bb = load_frag_bt(BtileT + nt * 16 * LDV_ + ks * 32, LDV_);
      acc[s] = wmma_bf16(a, bb, acc[s]);
    }
  }

  const int n_  = lane & 15;
  const int mb8 = (lane >> 4) << 3;
#pragma unroll
  for (int s = 0; s < 2; ++s) {
    int t  = wave + 8 * s;
    int mt = t >> 2, nt = t & 3;
#pragma unroll
    for (int r = 0; r < 8; ++r) {
      int o  = mtb * 64 + mt * 16 + mb8 + r;
      int hw = nb * 64 + nt * 16 + n_;
      size_t oi = ((size_t)b * C_ + o) * HW_ + hw;
      out[oi] = g * (acc[s][r] + bo[o]) + x[oi] + pos[(size_t)o * HW_ + hw];
    }
  }
}

// ---------------- host launcher ---------------------------------------------
extern "C" void kernel_launch(void* const* d_in, const int* in_sizes, int n_in,
                              void* d_out, int out_size, void* d_ws, size_t ws_size,
                              hipStream_t stream) {
  const float* x     = (const float*)d_in[0];
  const float* wq    = (const float*)d_in[1];
  const float* bq    = (const float*)d_in[2];
  const float* wk    = (const float*)d_in[3];
  const float* bk    = (const float*)d_in[4];
  const float* wv    = (const float*)d_in[5];
  const float* bv    = (const float*)d_in[6];
  const float* wo    = (const float*)d_in[7];
  const float* bo    = (const float*)d_in[8];
  const float* gamma = (const float*)d_in[9];
  float* out = (float*)d_out;

  char* ws = (char*)d_ws;
  size_t off = 0;
  auto alloc = [&](size_t bytes) -> void* {
    void* p = ws + off;
    off += (bytes + 255) & ~(size_t)255;
    return p;
  };
  float*  pos   = (float*)  alloc(sizeof(float) * C_ * HW_);                //  8 MB
  bf16_t* wqkvb = (bf16_t*) alloc(sizeof(bf16_t) * 3 * CH_ * C_);
  bf16_t* wob   = (bf16_t*) alloc(sizeof(bf16_t) * C_ * CH_);
  bf16_t* qkvb  = (bf16_t*) alloc(sizeof(bf16_t) * (size_t)B_ * 192 * HW_); // 12.6 MB
  bf16_t* outH  = (bf16_t*) alloc(sizeof(bf16_t) * (size_t)B_ * 64 * HW_);  //  4 MB
  bf16_t* outW  = (bf16_t*) alloc(sizeof(bf16_t) * (size_t)B_ * 64 * HW_);  //  4 MB
  (void)ws_size; (void)in_sizes; (void)n_in; (void)out_size;

  cca_pos_kernel<<<(C_ * HW_ + 255) / 256, 256, 0, stream>>>(pos);
  cca_wcvt_kernel<<<(3 * CH_ * C_ + C_ * CH_ + 255) / 256, 256, 0, stream>>>(
      wq, wk, wv, wo, wqkvb, wob);
  cca_qkv_kernel<<<B_ * 64, 256, 0, stream>>>(x, pos, wqkvb, bq, bk, bv, qkvb);
  cca_attn_kernel<<<B_ * NH_ * W_, 256, 0, stream>>>(qkvb, outH, outW);
  cca_out_kernel<<<B_ * 8 * 64, 256, 0, stream>>>(outH, outW, wob, bo, x, pos, gamma, out);
}
